// StructuredGNN_5574867550487
// MI455X (gfx1250) — matrix operand order repaired and checked by reading, not compiled, not run on hardware
//
#include <hip/hip_runtime.h>
#include <hip/hip_bf16.h>

typedef __attribute__((ext_vector_type(16))) __bf16 v16bf;
typedef __attribute__((ext_vector_type(8)))  float  v8f;

static constexpr int NN   = 50000;
static constexpr int EE   = 800000;
static constexpr int INF_ = 512;
static constexpr int HIDF = 64;
static constexpr int HDS  = 4;
static constexpr int NCF  = 40;

// ---------- helpers ----------
__device__ __forceinline__ __bf16 f2bf(float f) {
  unsigned u = __float_as_uint(f);
  unsigned r = (u + 0x7FFFu + ((u >> 16) & 1u)) >> 16;   // round-to-nearest-even
  unsigned short s = (unsigned short)r;
  return __builtin_bit_cast(__bf16, s);
}

__device__ __forceinline__ void atomicMaxF(float* addr, float val) {
  if (val >= 0.f) atomicMax((int*)addr, __float_as_int(val));
  else            atomicMin((unsigned int*)addr, __float_as_uint(val));
}

// ---------- B pre-swizzle: fp32 [K,ld] -> bf16 fragment layout ----------
// dst flat layout: [K/32][Ntiles][lane 0..31][16]
//   element (panel,tile,lane,i) = B[panel*32 + (lane>=16)*16 + i, tile*16 + lane%16]
__global__ void swizzle_B_kernel(const float* __restrict__ src, __bf16* __restrict__ dst,
                                 int K, int Ntiles, int ld, int Nsrc) {
  const long long total = (long long)(K >> 5) * Ntiles * 512;
  const long long idx = blockIdx.x * (long long)blockDim.x + threadIdx.x;
  if (idx >= total) return;
  const int i    = (int)(idx & 15);
  const int lane = (int)((idx >> 4) & 31);
  const long long rest = idx >> 9;
  const int tile  = (int)(rest % Ntiles);
  const int panel = (int)(rest / Ntiles);
  const int k = panel * 32 + ((lane >> 4) << 4) + i;
  const int n = tile * 16 + (lane & 15);
  const float v = (n < Nsrc) ? src[(size_t)k * ld + n] : 0.f;
  dst[idx] = f2bf(v);
}

// ---------- WMMA GEMM: C[M,Nstore] = act(A[M,K] @ B[K,16*Ntiles] + bias) ----------
// block = 128 threads (4 waves); block covers a 16-row M tile; wave w owns NT 16-col tiles.
// A (fp32) is cooperatively staged through LDS in bf16 fragment layout (shared by all waves).
// B comes pre-swizzled (bf16 fragment layout): 1 contiguous 32B load per lane per tile.
template <int NT>
__global__ __launch_bounds__(128) void wmma_gemm_kernel(
    const float* __restrict__ A, const __bf16* __restrict__ Bsw,
    const float* __restrict__ bias, float* __restrict__ C,
    int M, int Ntiles, int K, int ldc, int Nstore, int act) {
  __shared__ __align__(16) __bf16 ldsA[32][16];

  const int lane = threadIdx.x & 31;
  const int wave = threadIdx.x >> 5;
  const int m0 = blockIdx.x * 16;
  if (m0 >= M) return;                                   // block-uniform
  const int n_base = wave * (NT * 16);
  const bool active = (n_base >> 4) < Ntiles;            // wave-uniform

  v8f acc[NT];
#pragma unroll
  for (int t = 0; t < NT; ++t)
#pragma unroll
    for (int i = 0; i < 8; ++i) acc[t][i] = 0.f;

  // --- A staging map: thread t loads float4 at flat panel pos t*4 (row-major 16x32) ---
  const int sr  = threadIdx.x >> 3;                      // panel row 0..15
  const int skk = (threadIdx.x * 4) & 31;                // panel col 0,4,...,28
  const int slane = sr + (skk >> 4) * 16;                // fragment lane
  const int si    = skk & 15;                            // fragment element
  const float* arow = A + (size_t)(m0 + sr) * K + skk;

  for (int k0 = 0; k0 < K; k0 += 32) {
    const float4 f = *(const float4*)(arow + k0);
    if (k0 + 32 < K) __builtin_prefetch(arow + k0 + 32, 0, 3);
    __syncthreads();                                     // previous reads done
    {
      __bf16* d = &ldsA[slane][si];
      d[0] = f2bf(f.x); d[1] = f2bf(f.y); d[2] = f2bf(f.z); d[3] = f2bf(f.w);
    }
    __syncthreads();                                     // panel visible
    const v16bf a = *(const v16bf*)&ldsA[lane][0];
    if (active) {
#pragma unroll
      for (int t = 0; t < NT; ++t) {
        const int tg = (n_base >> 4) + t;
        const v16bf b = *(const v16bf*)(
            Bsw + ((((size_t)(k0 >> 5) * Ntiles + tg) * 32 + lane) << 4));
        acc[t] = __builtin_amdgcn_wmma_f32_16x16x32_bf16(
            /*neg_a=*/false, a, /*neg_b=*/false, b,
            /*c_mod=*/(short)0, acc[t], /*reuse_a=*/false, /*reuse_b=*/false);
      }
    }
  }

  if (!active) return;
  const int rbase = m0 + ((lane >> 4) << 3);
  const int col = lane & 15;
#pragma unroll
  for (int t = 0; t < NT; ++t) {
    const int c = n_base + t * 16 + col;
    if (c >= Nstore) continue;
    const float bv = bias ? bias[c] : 0.f;
#pragma unroll
    for (int i = 0; i < 8; ++i) {
      float v = acc[t][i] + bv;
      if (act == 1) v = v > 0.f ? v : 0.f;               // relu
      C[(size_t)(rbase + i) * ldc + c] = v;
    }
  }
}

// ---------- attention scores: a_src[n,h] = <h_row, att_src[h]>, same for dst ----------
__global__ void attn_scores_kernel(const float* __restrict__ h,
                                   const float* __restrict__ att_src,
                                   const float* __restrict__ att_dst,
                                   float* __restrict__ a_src, float* __restrict__ a_dst,
                                   int n, int H) {
  const int i = blockIdx.x * blockDim.x + threadIdx.x;
  if (i >= n * H) return;
  const int node = i / H, hd = i % H;
  const float4* row = (const float4*)(h + (size_t)node * H * 64 + hd * 64);
  const float4* as = (const float4*)(att_src + hd * 64);
  const float4* ad = (const float4*)(att_dst + hd * 64);
  float s = 0.f, d = 0.f;
#pragma unroll
  for (int c = 0; c < 16; ++c) {
    const float4 v = row[c], a1 = as[c], a2 = ad[c];
    s += v.x * a1.x + v.y * a1.y + v.z * a1.z + v.w * a1.w;
    d += v.x * a2.x + v.y * a2.y + v.z * a2.z + v.w * a2.w;
  }
  a_src[i] = s; a_dst[i] = d;
}

// ---------- fill ----------
__global__ void fill_f32_kernel(float* __restrict__ p, float v, long long count) {
  for (long long i = blockIdx.x * (long long)blockDim.x + threadIdx.x; i < count;
       i += (long long)gridDim.x * blockDim.x) p[i] = v;
}

// ---------- edge pass 1: e = leakyrelu(a_src[src]+a_dst[dst]); segment max ----------
__global__ void edge_max_kernel(const int* __restrict__ ei,
                                const float* __restrict__ a_src,
                                const float* __restrict__ a_dst,
                                float* __restrict__ ebuf, float* __restrict__ m,
                                int E, int H) {
  const long long total = (long long)E * H;
  for (long long i = blockIdx.x * (long long)blockDim.x + threadIdx.x; i < total;
       i += (long long)gridDim.x * blockDim.x) {
    const int e = (int)(i / H), hd = (int)(i % H);
    const int s = ei[e], d = ei[E + e];
    float v = a_src[(size_t)s * H + hd] + a_dst[(size_t)d * H + hd];
    v = v > 0.f ? v : 0.2f * v;
    ebuf[i] = v;
    atomicMaxF(&m[(size_t)d * H + hd], v);
  }
}

// ---------- edge pass 2: ex = exp(e - m[dst]); segment sum ----------
__global__ void edge_exp_kernel(const int* __restrict__ ei,
                                float* __restrict__ ebuf,
                                const float* __restrict__ m, float* __restrict__ den,
                                int E, int H) {
  const long long total = (long long)E * H;
  for (long long i = blockIdx.x * (long long)blockDim.x + threadIdx.x; i < total;
       i += (long long)gridDim.x * blockDim.x) {
    const int e = (int)(i / H), hd = (int)(i % H);
    const int d = ei[E + e];
    const float ex = __expf(ebuf[i] - m[(size_t)d * H + hd]);
    ebuf[i] = ex;
    atomicAdd(&den[(size_t)d * H + hd], ex);
  }
}

// ---------- edge pass 3: agg[dst] += h[src] * alpha  (wave per edge, C=64) ----------
__global__ void edge_agg_kernel(const int* __restrict__ ei,
                                const float* __restrict__ ebuf,
                                const float* __restrict__ den,
                                const float* __restrict__ hsrc,
                                float* __restrict__ agg, int E, int H) {
  const int HC = H * 64;
  const int gwave = (int)((blockIdx.x * (long long)blockDim.x + threadIdx.x) >> 5);
  const int lane = threadIdx.x & 31;
  const int nwaves = (int)(((long long)gridDim.x * blockDim.x) >> 5);
  for (int e = gwave; e < E; e += nwaves) {
    const int s = ei[e], d = ei[E + e];
    for (int f = lane; f < HC; f += 32) {
      const int hd = f >> 6;
      const float alpha = ebuf[(size_t)e * H + hd] /
                          (den[(size_t)d * H + hd] + 1e-16f);
      atomicAdd(&agg[(size_t)d * HC + f], hsrc[(size_t)s * HC + f] * alpha);
    }
  }
}

// ---------- bias + optional elu (in place) ----------
__global__ void bias_act_kernel(float* __restrict__ h, const float* __restrict__ bias,
                                long long count, int F, int act) {
  for (long long i = blockIdx.x * (long long)blockDim.x + threadIdx.x; i < count;
       i += (long long)gridDim.x * blockDim.x) {
    const int c = (int)(i % F);
    float v = h[i] + bias[c];
    if (act == 2) v = v > 0.f ? v : (__expf(v) - 1.f);   // elu
    h[i] = v;
  }
}

extern "C" void kernel_launch(void* const* d_in, const int* in_sizes, int n_in,
                              void* d_out, int out_size, void* d_ws, size_t ws_size,
                              hipStream_t stream) {
  (void)in_sizes; (void)n_in; (void)out_size; (void)ws_size;
  const float* x        = (const float*)d_in[0];
  const int*   ei       = (const int*)  d_in[1];
  const float* W1       = (const float*)d_in[2];
  const float* att_src1 = (const float*)d_in[3];
  const float* att_dst1 = (const float*)d_in[4];
  const float* b1       = (const float*)d_in[5];
  const float* W2       = (const float*)d_in[6];
  const float* att_src2 = (const float*)d_in[7];
  const float* att_dst2 = (const float*)d_in[8];
  const float* b2       = (const float*)d_in[9];
  const float* mw1      = (const float*)d_in[10];
  const float* mb1      = (const float*)d_in[11];
  const float* mw2      = (const float*)d_in[12];
  const float* mb2      = (const float*)d_in[13];
  float* out = (float*)d_out;

  // ---- workspace carve ----
  char* w = (char*)d_ws;
  auto alloc = [&](size_t bytes) -> void* {
    void* p = (void*)w;
    w += (bytes + 255) & ~(size_t)255;
    return p;
  };
  const size_t F1 = (size_t)HDS * HIDF;                       // 256
  float*  RA     = (float*)alloc((size_t)NN * F1 * 4);        // h1; later h2/agg2/hid
  float*  out1   = (float*)alloc((size_t)NN * F1 * 4);        // agg1 -> elu in place
  float*  ebuf   = (float*)alloc((size_t)EE * HDS * 4);       // e / ex (reused layer 2)
  float*  a_src1 = (float*)alloc((size_t)NN * HDS * 4);
  float*  a_dst1 = (float*)alloc((size_t)NN * HDS * 4);
  float*  m1     = (float*)alloc((size_t)NN * HDS * 4);
  float*  den1   = (float*)alloc((size_t)NN * HDS * 4);
  float*  a_src2 = (float*)alloc((size_t)NN * 4);
  float*  a_dst2 = (float*)alloc((size_t)NN * 4);
  float*  m2     = (float*)alloc((size_t)NN * 4);
  float*  den2   = (float*)alloc((size_t)NN * 4);
  __bf16* W1sw   = (__bf16*)alloc((size_t)INF_ * 256 * 2);    // [512/32][16][32][16]
  __bf16* W2sw   = (__bf16*)alloc((size_t)256 * 64 * 2);      // [256/32][4][32][16]
  __bf16* mw1sw  = (__bf16*)alloc((size_t)64 * 64 * 2);       // [2][4][32][16]
  __bf16* mw2sw  = (__bf16*)alloc((size_t)64 * 48 * 2);       // [2][3][32][16] (padded)
  float* h1   = RA;
  float* h2   = RA;                                           // aliases: h1 dead by then
  float* agg2 = RA + (size_t)NN * HIDF;
  float* hid  = RA + (size_t)2 * NN * HIDF;

  const float NEG = -3.402823466e38f;
  const int MT = NN / 16;                                     // 3125

  // ===== weight pre-swizzle (bf16 fragment layout) =====
  swizzle_B_kernel<<<512, 256, 0, stream>>>(W1,  W1sw,  INF_, 16, 256, 256);
  swizzle_B_kernel<<<128, 256, 0, stream>>>(W2,  W2sw,  256,   4,  64,  64);
  swizzle_B_kernel<<< 32, 256, 0, stream>>>(mw1, mw1sw,  64,   4,  64,  64);
  swizzle_B_kernel<<< 24, 256, 0, stream>>>(mw2, mw2sw,  64,   3,  40,  40);

  // ===== Layer 1 =====
  // h1 = x @ W1   [50000,512]x[512,256]
  wmma_gemm_kernel<4><<<MT, 128, 0, stream>>>(x, W1sw, nullptr, h1,
      NN, 16, INF_, 256, 256, 0);
  attn_scores_kernel<<<(NN * HDS + 255) / 256, 256, 0, stream>>>(
      h1, att_src1, att_dst1, a_src1, a_dst1, NN, HDS);
  fill_f32_kernel<<<1024, 256, 0, stream>>>(m1, NEG, (long long)NN * HDS);
  fill_f32_kernel<<<1024, 256, 0, stream>>>(den1, 0.f, (long long)NN * HDS);
  fill_f32_kernel<<<16384, 256, 0, stream>>>(out1, 0.f, (long long)NN * F1);
  edge_max_kernel<<<12500, 256, 0, stream>>>(ei, a_src1, a_dst1, ebuf, m1, EE, HDS);
  edge_exp_kernel<<<12500, 256, 0, stream>>>(ei, ebuf, m1, den1, EE, HDS);
  edge_agg_kernel<<<100000, 256, 0, stream>>>(ei, ebuf, den1, h1, out1, EE, HDS);
  bias_act_kernel<<<16384, 256, 0, stream>>>(out1, b1, (long long)NN * F1, (int)F1, 2);

  // ===== Layer 2 (heads = 1) =====
  // h2 = out1 @ W2   [50000,256]x[256,64]
  wmma_gemm_kernel<1><<<MT, 128, 0, stream>>>(out1, W2sw, nullptr, h2,
      NN, 4, (int)F1, 64, 64, 0);
  attn_scores_kernel<<<(NN + 255) / 256, 256, 0, stream>>>(
      h2, att_src2, att_dst2, a_src2, a_dst2, NN, 1);
  fill_f32_kernel<<<1024, 256, 0, stream>>>(m2, NEG, (long long)NN);
  fill_f32_kernel<<<1024, 256, 0, stream>>>(den2, 0.f, (long long)NN);
  fill_f32_kernel<<<8192, 256, 0, stream>>>(agg2, 0.f, (long long)NN * HIDF);
  edge_max_kernel<<<3125, 256, 0, stream>>>(ei, a_src2, a_dst2, ebuf, m2, EE, 1);
  edge_exp_kernel<<<3125, 256, 0, stream>>>(ei, ebuf, m2, den2, EE, 1);
  edge_agg_kernel<<<100000, 256, 0, stream>>>(ei, ebuf, den2, h2, agg2, EE, 1);
  bias_act_kernel<<<8192, 256, 0, stream>>>(agg2, b2, (long long)NN * HIDF, HIDF, 0);

  // ===== MLP head =====
  // hid = relu(agg2 @ mw1 + mb1)   [50000,64]x[64,64]
  wmma_gemm_kernel<1><<<MT, 128, 0, stream>>>(agg2, mw1sw, mb1, hid,
      NN, 4, 64, 64, 64, 1);
  // out = hid @ mw2sw + mb2   [50000,64]x[64,48->40]
  wmma_gemm_kernel<1><<<MT, 128, 0, stream>>>(hid, mw2sw, mb2, out,
      NN, 3, 64, NCF, NCF, 0);
}
